// minerva2_35124242547416
// MI455X (gfx1250) — matrix operand order repaired and checked by reading, not compiled, not run on hardware
//
#include <hip/hip_runtime.h>
#include <hip/hip_bf16.h>

typedef __attribute__((ext_vector_type(8)))  unsigned short v8u16;
typedef __attribute__((ext_vector_type(16))) unsigned short v16u16;
typedef __attribute__((ext_vector_type(16))) __bf16        v16bf;
typedef __attribute__((ext_vector_type(8)))  float         v8f;

// ---- helpers -------------------------------------------------------------

static __device__ __forceinline__ unsigned short f2bf(float f) {
  // round-to-nearest-even f32 -> bf16
  unsigned int u = __builtin_bit_cast(unsigned int, f);
  unsigned int r = u + 0x7FFFu + ((u >> 16) & 1u);
  if ((u & 0x7F800000u) == 0x7F800000u) r = u;  // inf/nan passthrough
  return (unsigned short)(r >> 16);
}

static __device__ __forceinline__ v16bf mkbf(v8u16 lo, v8u16 hi) {
  v16u16 v = __builtin_shufflevector(lo, hi, 0,1,2,3,4,5,6,7,8,9,10,11,12,13,14,15);
  return __builtin_bit_cast(v16bf, v);
}

static __device__ __forceinline__ v16bf ldbf32(const unsigned short* p) {
  return __builtin_bit_cast(v16bf, *(const v16u16*)p);
}

static __device__ __forceinline__ v8f wmma_bf16(v16bf a, v16bf b, v8f c) {
  return __builtin_amdgcn_wmma_f32_16x16x32_bf16(false, a, false, b, (short)0, c, false, false);
}

// ---- phase 1: row L2-normalize + f32 -> bf16 -----------------------------
// one wave per row of 256 floats; 8 rows per 256-thread block
__global__ __launch_bounds__(256) void minerva2_nrm_kernel(
    const float* __restrict__ in, unsigned short* __restrict__ outb, int rows) {
  int lane = threadIdx.x & 31;
  int row  = blockIdx.x * 8 + (threadIdx.x >> 5);
  if (row >= rows) return;
  const float* r = in + (size_t)row * 256;
  float x[8];
  float ss = 0.f;
#pragma unroll
  for (int j = 0; j < 8; ++j) { x[j] = r[j * 32 + lane]; ss += x[j] * x[j]; }
#pragma unroll
  for (int m = 1; m < 32; m <<= 1) ss += __shfl_xor(ss, m);
  float scale = 1.0f / fmaxf(sqrtf(ss), 1e-12f);
  unsigned short* o = outb + (size_t)row * 256;
#pragma unroll
  for (int j = 0; j < 8; ++j) o[j * 32 + lane] = f2bf(x[j] * scale);
}

// ---- phase 2: pre-swizzle ex_class_reps into per-lane WMMA B fragments ---
// repsB[(g*32 + lane)*32 + {0..15 = tile c0..15, 16..31 = tile c16..31}]
// B layout (32Kx16N bf16): lane l holds column c=l&15, K = 8*(l>>4)+j (j<8), 16+8*(l>>4)+(j-8) (j>=8)
__global__ __launch_bounds__(256) void minerva2_packreps_kernel(
    const float* __restrict__ reps, unsigned short* __restrict__ repsB, int D, int C) {
  int t = blockIdx.x * blockDim.x + threadIdx.x;  // one thread per (chunk, lane)
  if (t >= D) return;
  int lane = t & 31, g = t >> 5;
  int lh = lane >> 4, ll = lane & 15;
  unsigned short* o = repsB + (size_t)t * 32;
#pragma unroll
  for (int j = 0; j < 16; ++j) {
    int d = g * 32 + (j < 8 ? lh * 8 + j : 16 + lh * 8 + (j - 8));
    o[j] = f2bf(reps[(size_t)d * C + ll]);          // c = ll < 16 <= C always valid
    int c1 = 16 + ll;
    o[16 + j] = (c1 < C) ? f2bf(reps[(size_t)d * C + c1]) : (unsigned short)0;
  }
}

// ---- phase 3: fused  s = fn@en^T ; a = sign(s)|s|^p ; echo += a@reps ; intens += rowsum(a)
// 8 waves/block, wave owns 16 N-rows, block owns 128 N-rows; grid.y splits D.
// en chunks are staged block-wide into LDS with async-to-LDS loads, double-buffered.
__global__ __launch_bounds__(256) void minerva2_fused_kernel(
    const unsigned short* __restrict__ fnb,    // [N][256] bf16, normalized
    const unsigned short* __restrict__ enb,    // [D][256] bf16, normalized
    const unsigned short* __restrict__ repsB,  // [D/32][32][32] bf16, B-fragments
    const int* __restrict__ pfac,
    float* __restrict__ out, int N, int D, int dPerBlock) {
  // en staging: 32 d-rows x 256 halves, padded to 264 (528B row stride:
  // 16B-aligned for b128, and lanes 0..15 land on distinct bank quads)
  __shared__ __align__(16) unsigned short enbuf[2][32][264];
  __shared__ __align__(16) unsigned short a_lds[8][16][40];  // per-wave transpose slice

  const int lane = threadIdx.x & 31;
  const int wave = threadIdx.x >> 5;
  const int lh = lane >> 4;    // half of wave
  const int ll = lane & 15;
  const int n0 = blockIdx.x * 128 + wave * 16;
  const int d_begin = blockIdx.y * dPerBlock;
  const int d_end = d_begin + dPerBlock;
  const int p = *pfac;

  // cooperative async stage of en[dbase..dbase+31][0..255] into enbuf[buf]
  auto stage = [&](int dbase, int buf) {
    const int t = threadIdx.x;
#pragma unroll
    for (int i = 0; i < 4; ++i) {
      int piece = t + 256 * i;      // 1024 x 16B pieces per chunk
      int row = piece >> 5;         // d-row within chunk
      int col = piece & 31;         // 16B piece within row
      unsigned lds = (unsigned)(uintptr_t)&enbuf[buf][row][col * 8];
      const unsigned short* gp = enb + (size_t)(dbase + row) * 256 + col * 8;
      asm volatile("global_load_async_to_lds_b128 %0, %1, off"
                   :: "v"(lds), "v"((unsigned long long)(uintptr_t)gp)
                   : "memory");
    }
  };

  // A fragments: 16 rows x K=256, resident for the whole D loop.
  // A 16x32 bf16 layout: lane holds M=ll, K = 8*lh + {0..7} and 16 + 8*lh + {0..7}
  v16bf A[8];
  {
    const unsigned short* arow = fnb + (size_t)(n0 + ll) * 256 + 8 * lh;
#pragma unroll
    for (int k = 0; k < 8; ++k) {
      v8u16 lo = *(const v8u16*)(arow + 32 * k);
      v8u16 hi = *(const v8u16*)(arow + 32 * k + 16);
      A[k] = mkbf(lo, hi);
    }
  }

  v8f echo0 = {}, echo1 = {};
  float intens[8] = {0.f, 0.f, 0.f, 0.f, 0.f, 0.f, 0.f, 0.f};

  // prologue: fill buffer 0
  int b = 0;
  stage(d_begin, 0);
  asm volatile("s_wait_asynccnt 0" ::: "memory");
  __syncthreads();

  for (int d0 = d_begin; d0 < d_end; d0 += 32) {
    // pipeline: start filling the other buffer (it was fully consumed before
    // the barrier that ended the previous iteration)
    if (d0 + 32 < d_end) stage(d0 + 32, b ^ 1);

    // ---- main GEMM: two 16x16 s tiles (cols d0..+15 / d0+16..+31) from LDS
    const unsigned short* bxp = &enbuf[b][ll][8 * lh];
    const unsigned short* byp = &enbuf[b][16 + ll][8 * lh];
    v8f sX = {}, sY = {};
#pragma unroll
    for (int k = 0; k < 8; ++k) {
      v16bf Bx = mkbf(*(const v8u16*)(bxp + 32 * k), *(const v8u16*)(bxp + 32 * k + 16));
      v16bf By = mkbf(*(const v8u16*)(byp + 32 * k), *(const v8u16*)(byp + 32 * k + 16));
      sX = wmma_bf16(A[k], Bx, sX);
      sY = wmma_bf16(A[k], By, sY);
    }

    // ---- activation a = sign(s)|s|^p, intensity accumulation, spill a->LDS (bf16)
    // C layout: acc[i] is (row = i + 8*lh, col = ll [+16 for Y])
#pragma unroll
    for (int i = 0; i < 8; ++i) {
      float ax, ay;
      if (p == 3) {
        ax = sX[i] * sX[i] * sX[i];
        ay = sY[i] * sY[i] * sY[i];
      } else {
        float mx = powf(fabsf(sX[i]), (float)p);
        float my = powf(fabsf(sY[i]), (float)p);
        ax = (sX[i] < 0.f) ? -mx : mx;
        ay = (sY[i] < 0.f) ? -my : my;
      }
      intens[i] += ax + ay;
      int row = i + 8 * lh;
      a_lds[wave][row][ll]      = f2bf(ax);
      a_lds[wave][row][16 + ll] = f2bf(ay);
    }
    // wave-private LDS slice; DS ops are in-order per wave — just drain DScnt
    asm volatile("s_wait_dscnt 0" ::: "memory");

    // ---- reload a in A-fragment layout (the LDS round trip is the transpose)
    const unsigned short* a2p = &a_lds[wave][ll][0];
    v16bf A2 = mkbf(*(const v8u16*)(a2p + 8 * lh), *(const v8u16*)(a2p + 16 + 8 * lh));

    // ---- echo GEMM: echo += a(16x32) @ reps(32x32 padded)
    const unsigned short* rp = repsB + ((size_t)(d0 >> 5) * 32 + lane) * 32;
    echo0 = wmma_bf16(A2, ldbf32(rp), echo0);
    echo1 = wmma_bf16(A2, ldbf32(rp + 16), echo1);

    // retire this wave's async fills, then block-wide sync before swapping
    asm volatile("s_wait_asynccnt 0" ::: "memory");
    __syncthreads();
    b ^= 1;
  }

  // ---- writeback: echo (f32 atomics; grid.y partitions D)
  const int C = 28;
  const int ECHO = N * C;
#pragma unroll
  for (int i = 0; i < 8; ++i) {
    int n = n0 + i + 8 * lh;
    atomicAdd(&out[(size_t)n * C + ll], echo0[i]);                        // cols 0..15
    if (ll < C - 16) atomicAdd(&out[(size_t)n * C + 16 + ll], echo1[i]);  // cols 16..27
  }
  // ---- intensity: reduce across the 16 d-columns held by each half-wave
#pragma unroll
  for (int i = 0; i < 8; ++i) {
    float v = intens[i];
    v += __shfl_xor(v, 1);
    v += __shfl_xor(v, 2);
    v += __shfl_xor(v, 4);
    v += __shfl_xor(v, 8);
    if (ll == 0) atomicAdd(&out[ECHO + n0 + i + 8 * lh], v);
  }
}

// ---- launch --------------------------------------------------------------

extern "C" void kernel_launch(void* const* d_in, const int* in_sizes, int n_in,
                              void* d_out, int out_size, void* d_ws, size_t ws_size,
                              hipStream_t stream) {
  const float* features    = (const float*)d_in[0];
  const float* ex_features = (const float*)d_in[1];
  const float* ex_reps     = (const float*)d_in[2];
  const int*   pfac        = (const int*)d_in[3];
  float* out = (float*)d_out;

  const int H = 256;
  const int N = in_sizes[0] / H;       // 4096
  const int D = in_sizes[1] / H;       // 16384
  const int C = in_sizes[2] / D;       // 28

  unsigned short* fnb   = (unsigned short*)d_ws;          //  N*H bf16 (2 MB)
  unsigned short* enb   = fnb + (size_t)N * H;            //  D*H bf16 (8 MB)
  unsigned short* repsB = enb + (size_t)D * H;            //  D*32 bf16 (1 MB)

  hipMemsetAsync(d_out, 0, (size_t)out_size * sizeof(float), stream);

  minerva2_nrm_kernel<<<N / 8, 256, 0, stream>>>(features, fnb, N);
  minerva2_nrm_kernel<<<D / 8, 256, 0, stream>>>(ex_features, enb, D);
  minerva2_packreps_kernel<<<(D + 255) / 256, 256, 0, stream>>>(ex_reps, repsB, D, C);

  const int DSPLIT = 8;                 // 32 x 8 = 256 workgroups
  dim3 grid(N / 128, DSPLIT);
  minerva2_fused_kernel<<<grid, 256, 0, stream>>>(fnb, enb, repsB, pfac, out,
                                                  N, D, D / DSPLIT);
}
